// HqqInt4Linear_52072183497435
// MI455X (gfx1250) — compile-verified, matrix operand order checked by reading, not run on hardware
//
#include <hip/hip_runtime.h>
#include <hip/hip_bf16.h>

// ---------------------------------------------------------------------------
// HQQ int4 linear for MI455X (gfx1250), wave32 + WMMA bf16 + async LDS DMA.
//   out[4096 x 4096] = X[4096 x 4096] * W[4096 x 4096]^T
//
// Pipeline:
//   k_cvt_x : X f32 -> bf16 into ws[0 .. 32MB)
//   k_deq_w : packed int4 -> bf16 W into ws[32MB .. 64MB)
//   k_gemm  : 128x128x32 tiles, TRIPLE-buffered LDS fed by
//             global_load_async_to_lds_b128 (ASYNCcnt, in-order completion,
//             so s_wait_asynccnt 4 drains only the oldest stage),
//             v_wmma_f32_16x16x32_bf16 compute.
// ---------------------------------------------------------------------------

typedef __attribute__((ext_vector_type(16))) __bf16 bf16x16;
typedef __attribute__((ext_vector_type(8)))  float  f32x8;
typedef int i32x4v __attribute__((vector_size(16)));   // matches builtin's pointee

union Bf16Frag {
    bf16x16 v;
    uint4   q[2];
};

#define TOKENS      4096      // 2 * 2048
#define OUTF        4096
#define INF         4096
#define PACKED_N    8388608   // OUTF*INF/2 bytes
#define XBF_ELEMS   16777216  // TOKENS*INF
#define BM          128
#define BN          128
#define BK          32
#define LDA         40        // LDS row stride in bf16 elems (80B, pads banks)
#define NSTAGE      3

// Typed, address-space-qualified pointer casts for the async-DMA builtin:
// src: global (AS1) int4*, dst: LDS (AS3) int4*.
#define ASG(p) ((__attribute__((address_space(1))) i32x4v*)(p))
#define ASL(p) ((__attribute__((address_space(3))) i32x4v*)(p))

static __device__ __forceinline__ unsigned short f32_to_bf16(float f) {
    unsigned u = __float_as_uint(f);
    unsigned r = (u + 0x7FFFu + ((u >> 16) & 1u)) >> 16;   // round-nearest-even
    return (unsigned short)r;
}

// ---------------------------------------------------------------------------
// Kernel 1: X f32 -> bf16 (4 elems / thread)
// ---------------------------------------------------------------------------
__global__ __launch_bounds__(256) void k_cvt_x(const float* __restrict__ x,
                                               unsigned short* __restrict__ xbf) {
    int t = blockIdx.x * 256 + threadIdx.x;          // 0 .. 4194303
    float4 v = reinterpret_cast<const float4*>(x)[t];
    reinterpret_cast<ushort4*>(xbf)[t] =
        make_ushort4(f32_to_bf16(v.x), f32_to_bf16(v.y),
                     f32_to_bf16(v.z), f32_to_bf16(v.w));
}

// ---------------------------------------------------------------------------
// Kernel 2: dequantize packed int4 -> bf16 W [OUTF x INF] row-major.
// Concat layout: element flat j<8M is HIGH nibble of byte j, flat j+8M is
// LOW nibble of byte j. Group id = flat >> 6. 4 bytes / thread.
// ---------------------------------------------------------------------------
__global__ __launch_bounds__(256) void k_deq_w(const unsigned char* __restrict__ wq,
                                               const float* __restrict__ scale,
                                               const float* __restrict__ zero,
                                               unsigned short* __restrict__ wbf) {
    int t = blockIdx.x * 256 + threadIdx.x;          // 0 .. 2097151
    unsigned bytes = reinterpret_cast<const unsigned*>(wq)[t];

    int gh = t >> 4;                                 // (t*4) >> 6, same for all 4 bytes
    int gl = gh + (PACKED_N >> 6);                   // +131072
    float sh = scale[gh], zh = zero[gh];
    float sl = scale[gl], zl = zero[gl];

    unsigned short h[4], l[4];
#pragma unroll
    for (int b = 0; b < 4; ++b) {
        unsigned byte = (bytes >> (8 * b)) & 0xFFu;
        h[b] = f32_to_bf16(((float)(byte >> 4)   - zh) * sh);
        l[b] = f32_to_bf16(((float)(byte & 15u)  - zl) * sl);
    }
    reinterpret_cast<ushort4*>(wbf)[t] = make_ushort4(h[0], h[1], h[2], h[3]);
    reinterpret_cast<ushort4*>(wbf + PACKED_N)[t] = make_ushort4(l[0], l[1], l[2], l[3]);
}

// ---------------------------------------------------------------------------
// Kernel 3: bf16 GEMM via v_wmma_f32_16x16x32_bf16.
// Block = 256 threads = 8 waves (4 along M x 2 along N).
// Wave tile = 32(M) x 64(N) = 2x4 WMMA accumulators.
// 3-stage async pipeline: at iter kt, DMA stage kt+2 and wait only for the
// oldest 4 outstanding ops (stage kt+1) via s_wait_asynccnt 4.
// ---------------------------------------------------------------------------
__global__ __launch_bounds__(256) void k_gemm(const unsigned short* __restrict__ xbf,
                                              const unsigned short* __restrict__ wbf,
                                              float* __restrict__ out) {
    __shared__ unsigned short sA[NSTAGE][BM * LDA];   // 3 x 10240 B
    __shared__ unsigned short sB[NSTAGE][BN * LDA];   // 3 x 10240 B

    const int tid    = threadIdx.x;
    const int lane   = tid & 31;
    const int wave   = tid >> 5;        // 0..7
    const int waveM  = wave >> 1;       // 0..3  (32 rows each)
    const int waveN  = wave & 1;        // 0..1  (64 cols each)
    const int lane16 = lane & 15;
    const int kb     = (lane >> 4) * 8; // 0 for lanes 0-15, 8 for lanes 16-31

    const int bM = blockIdx.y * BM;
    const int bN = blockIdx.x * BN;

    // Global->LDS chunk mapping: 512 chunks of 8 bf16 per 128x32 tile,
    // 2 chunks per thread per tile.
    const int c0  = tid;
    const int c1  = tid + 256;
    const int ar0 = c0 >> 2, ac0 = (c0 & 3) * 8;
    const int ar1 = c1 >> 2, ac1 = (c1 & 3) * 8;

    const unsigned short* Ap = xbf + (size_t)bM * INF;
    const unsigned short* Bp = wbf + (size_t)bN * INF;

    // DMA one 128x32 A slab + one 128x32 B slab into LDS stage s (4 ops/thread).
    auto issue_stage = [&](int s, int k0) {
        __builtin_amdgcn_global_load_async_to_lds_b128(ASG(Ap + ar0 * INF + k0 + ac0),
                                                       ASL(&sA[s][ar0 * LDA + ac0]), 0, 0);
        __builtin_amdgcn_global_load_async_to_lds_b128(ASG(Ap + ar1 * INF + k0 + ac1),
                                                       ASL(&sA[s][ar1 * LDA + ac1]), 0, 0);
        __builtin_amdgcn_global_load_async_to_lds_b128(ASG(Bp + ar0 * INF + k0 + ac0),
                                                       ASL(&sB[s][ar0 * LDA + ac0]), 0, 0);
        __builtin_amdgcn_global_load_async_to_lds_b128(ASG(Bp + ar1 * INF + k0 + ac1),
                                                       ASL(&sB[s][ar1 * LDA + ac1]), 0, 0);
    };

    f32x8 acc[2][4] = {};

    const int KT = INF / BK;            // 128

    // ---- prologue: stages 0 and 1 in flight; drain stage 0 only ----
    issue_stage(0, 0);
    issue_stage(1, BK);
    __builtin_amdgcn_s_wait_asynccnt(4);
    __syncthreads();

    int stage = 0;
    for (int kt = 0; kt < KT; ++kt) {
        // DMA stage kt+2 (its buffer's last readers finished at iter kt-1,
        // behind that iteration's barrier).
        if (kt + 2 < KT) {
            int s2 = stage + 2; if (s2 >= NSTAGE) s2 -= NSTAGE;
            issue_stage(s2, (kt + 2) * BK);
        }

        // ---- A fragments (ISA 16-bit A layout: halves at kb / kb+16) ----
        const int rA = waveM * 32 + lane16;
        Bf16Frag af0, af1;
        af0.q[0] = *reinterpret_cast<const uint4*>(&sA[stage][rA * LDA + kb]);
        af0.q[1] = *reinterpret_cast<const uint4*>(&sA[stage][rA * LDA + kb + 16]);
        af1.q[0] = *reinterpret_cast<const uint4*>(&sA[stage][(rA + 16) * LDA + kb]);
        af1.q[1] = *reinterpret_cast<const uint4*>(&sA[stage][(rA + 16) * LDA + kb + 16]);

        // ---- B fragments interleaved with WMMAs (lower register pressure) ----
#pragma unroll
        for (int tn = 0; tn < 4; ++tn) {
            const int rB = waveN * 64 + tn * 16 + lane16;
            Bf16Frag bf;
            bf.q[0] = *reinterpret_cast<const uint4*>(&sB[stage][rB * LDA + kb]);
            bf.q[1] = *reinterpret_cast<const uint4*>(&sB[stage][rB * LDA + kb + 16]);
            acc[0][tn] = __builtin_amdgcn_wmma_f32_16x16x32_bf16(
                false, af0.v, false, bf.v, (short)0, acc[0][tn], false, false);
            acc[1][tn] = __builtin_amdgcn_wmma_f32_16x16x32_bf16(
                false, af1.v, false, bf.v, (short)0, acc[1][tn], false, false);
        }

        // ---- drain the stage we compute next; keep the one after in flight ----
        if (kt + 2 < KT) {
            __builtin_amdgcn_s_wait_asynccnt(4);
        } else if (kt + 1 < KT) {
            __builtin_amdgcn_s_wait_asynccnt(0);
        }
        if (kt + 1 < KT) {
            __syncthreads();
            ++stage; if (stage == NSTAGE) stage = 0;
        }
    }

    // ---- epilogue: C/D layout: VGPR r -> M=r (lanes 0-15) / M=r+8 (16-31) ----
#pragma unroll
    for (int tm = 0; tm < 2; ++tm) {
        int rowBase = bM + waveM * 32 + tm * 16 + kb;
#pragma unroll
        for (int tn = 0; tn < 4; ++tn) {
            int col = bN + waveN * 64 + tn * 16 + lane16;
#pragma unroll
            for (int r = 0; r < 8; ++r)
                out[(size_t)(rowBase + r) * OUTF + col] = acc[tm][tn][r];
        }
    }
}

// ---------------------------------------------------------------------------
// Host-side launcher. Inputs: x (f32), W_q (u8), scale (f32), zero (f32).
// Workspace: [0,32MB) X bf16, [32MB,64MB) W bf16.
// ---------------------------------------------------------------------------
extern "C" void kernel_launch(void* const* d_in, const int* in_sizes, int n_in,
                              void* d_out, int out_size, void* d_ws, size_t ws_size,
                              hipStream_t stream) {
    const float*         x     = (const float*)d_in[0];
    const unsigned char* wq    = (const unsigned char*)d_in[1];
    const float*         scale = (const float*)d_in[2];
    const float*         zero  = (const float*)d_in[3];
    float*               out   = (float*)d_out;

    unsigned short* xbf = (unsigned short*)d_ws;            // 32 MB
    unsigned short* wbf = xbf + XBF_ELEMS;                  // next 32 MB

    k_cvt_x<<<XBF_ELEMS / (256 * 4), 256, 0, stream>>>(x, xbf);
    k_deq_w<<<(PACKED_N / 4) / 256, 256, 0, stream>>>(wq, scale, zero, wbf);

    dim3 grid(OUTF / BN, TOKENS / BM);   // 32 x 32
    k_gemm<<<grid, 256, 0, stream>>>(xbf, wbf, out);
}
